// MocFFn_63857573757195
// MI455X (gfx1250) — matrix operand order repaired and checked by reading, not compiled, not run on hardware
//
#include <hip/hip_runtime.h>

// ---------------------------------------------------------------------------
// MocFFn for MI455X (gfx1250): bf16 WMMA pipeline, register-blocked, with
// async-to-LDS staging of the activation tile (double buffered, ASYNCcnt).
//   N = B*S = 4096 tokens, D = 2048, H = 8192, TOPK = 1024
// Workspace (~240 MB of d_ws): xb, wub, wgb, wdb (bf16), gb, tb (bf16), thr.
// All bf16 operands (112 MB) fit in the 192 MB L2, so tile re-reads are L2
// hits; HBM traffic is first-touch fp32 reads + intermediates (~0.5 GB).
// ---------------------------------------------------------------------------

#define D_MODEL 2048
#define HIDDEN  8192
#define NTOK    4096
#define TOPK    1024

typedef __bf16 v8bf  __attribute__((ext_vector_type(8)));
typedef __bf16 v16bf __attribute__((ext_vector_type(16)));
typedef float  v8f   __attribute__((ext_vector_type(8)));

__device__ __forceinline__ unsigned short f2bf(float f) {
  unsigned int u = __float_as_uint(f);
  unsigned int r = u + 0x7FFFu + ((u >> 16) & 1u);   // round-to-nearest-even
  return (unsigned short)(r >> 16);
}

__device__ __forceinline__ float bf2f(unsigned short u) {
  return __uint_as_float(((unsigned int)u) << 16);
}

// Build a 16xbf16 WMMA fragment from two 16-byte chunks (global or LDS).
__device__ __forceinline__ v16bf load_frag(const unsigned short* base, int off0, int off1) {
  v8bf lo = *(const v8bf*)(base + off0);
  v8bf hi = *(const v8bf*)(base + off1);
  return __builtin_shufflevector(lo, hi, 0, 1, 2, 3, 4, 5, 6, 7,
                                         8, 9, 10, 11, 12, 13, 14, 15);
}

// Stage a 64x32 bf16 tile (4 KB) of `src` rows [m0,m0+64) x k [k0,k0+32)
// into LDS with gfx1250 async-to-LDS loads (one b128 per thread, 256 thr).
__device__ __forceinline__ void stage_a_async(
    const unsigned short* __restrict__ src, int ldk, int m0, int k0,
    unsigned short* sbuf) {
  const int t   = threadIdx.x;
  const int row = t >> 2;   // 0..63
  const int ch  = t & 3;    // four 16B chunks per 64B row
  unsigned long long ga =
      (unsigned long long)(const void*)(src + (size_t)(m0 + row) * ldk + k0 + ch * 8);
  unsigned int lds_off = (unsigned int)(unsigned long long)(void*)(sbuf + row * 32 + ch * 8);
  asm volatile("global_load_async_to_lds_b128 %0, %1, off"
               :: "v"(lds_off), "v"(ga) : "memory");
}

__device__ __forceinline__ void wait_async0() {
  asm volatile("s_wait_asynccnt 0" ::: "memory");
}

// ------------------------- fp32 -> bf16 conversion -------------------------
__global__ __launch_bounds__(256) void cvt_bf16_kernel(
    const float* __restrict__ src, unsigned short* __restrict__ dst, long long count) {
  long long i = (long long)blockIdx.x * blockDim.x + threadIdx.x;
  long long stride = (long long)gridDim.x * blockDim.x;
  for (; i < count; i += stride) dst[i] = f2bf(src[i]);
}

// ------------------- fused up + gate GEMM (bf16 WMMA) ----------------------
// Grid: (H/128, N/64), block 256 (8 waves).
// Block tile: 64 tokens x 128 h. Wave tile: 64 tokens x 16 h (4 sub-tiles),
// two accumulators per sub-tile (up, gate). A staged in LDS via async loads.
__global__ __launch_bounds__(256) void up_gate_kernel(
    const unsigned short* __restrict__ xb,
    const unsigned short* __restrict__ wub,
    const unsigned short* __restrict__ wgb,
    unsigned short* __restrict__ gb,
    unsigned short* __restrict__ tb) {
  __shared__ __align__(16) unsigned short sA[2][64 * 32];

  const int lane   = threadIdx.x & 31;
  const int wave   = threadIdx.x >> 5;
  const int m0     = blockIdx.y << 6;
  const int h0     = (blockIdx.x << 7) + (wave << 4);
  const int r      = lane & 15;
  const int hiHalf = lane >> 4;

  // B (32x16): lane<16 holds K{0..15} of column h0+r; lane>=16 holds K{16..31}
  const unsigned short* buBase = wub + (size_t)(h0 + r) * D_MODEL + (hiHalf ? 16 : 0);
  const unsigned short* bgBase = wgb + (size_t)(h0 + r) * D_MODEL + (hiHalf ? 16 : 0);
  const int aoff = hiHalf ? 16 : 0;  // A: K{0..15} vs K{16..31}

  v8f cu[4] = {};
  v8f cg[4] = {};

  stage_a_async(xb, D_MODEL, m0, 0, sA[0]);

  const int KT = D_MODEL / 32;
  for (int kt = 0; kt < KT; ++kt) {
    wait_async0();          // own tile-kt writes landed
    __syncthreads();        // everyone's tile-kt landed; prior reads of other buf done
    if (kt + 1 < KT) stage_a_async(xb, D_MODEL, m0, (kt + 1) * 32, sA[(kt + 1) & 1]);

    const int k0 = kt * 32;
    v16bf bu = load_frag(buBase + k0, 0, 8);
    v16bf bg = load_frag(bgBase + k0, 0, 8);
    const unsigned short* sa = sA[kt & 1];
#pragma unroll
    for (int tt = 0; tt < 4; ++tt) {
      v16bf a = load_frag(sa + (tt * 16 + r) * 32 + aoff, 0, 8);
      cu[tt] = __builtin_amdgcn_wmma_f32_16x16x32_bf16(false, a, false, bu, (short)0, cu[tt], false, false);
      cg[tt] = __builtin_amdgcn_wmma_f32_16x16x32_bf16(false, a, false, bg, (short)0, cg[tt], false, false);
    }
  }

  // C/D layout: VGPR i, lane<16 -> (M=i, N=lane); lane>=16 -> (M=i+8, N=lane-16)
  const int col = h0 + r;
#pragma unroll
  for (int tt = 0; tt < 4; ++tt) {
#pragma unroll
    for (int i = 0; i < 8; ++i) {
      const int row = m0 + tt * 16 + i + (hiHalf ? 8 : 0);
      const float gv = cg[tt][i];
      const float hv = cu[tt][i];
      const float act = gv / (1.0f + __expf(-gv));   // silu(g)
      gb[(size_t)row * HIDDEN + col] = f2bf(gv);
      tb[(size_t)row * HIDDEN + col] = f2bf(hv * act);
    }
  }
}

// ----------------- per-token top-k threshold (radix select) ----------------
__global__ __launch_bounds__(256) void topk_thresh_kernel(
    const unsigned short* __restrict__ gb, float* __restrict__ thr) {
  __shared__ unsigned int hist[256];
  __shared__ int s_hi, s_krem;
  const int n   = blockIdx.x;
  const int tid = threadIdx.x;
  const unsigned short* row = gb + (size_t)n * HIDDEN;

  hist[tid] = 0;
  __syncthreads();
  for (int i = tid; i < HIDDEN; i += 256) {
    unsigned int u = row[i];
    unsigned int key = u ^ ((u & 0x8000u) ? 0xFFFFu : 0x8000u);  // monotone key
    atomicAdd(&hist[key >> 8], 1u);
  }
  __syncthreads();
  if (tid == 0) {
    int cum = 0, b = 255;
    for (; b > 0; --b) {
      int c = (int)hist[b];
      if (cum + c >= TOPK) break;
      cum += c;
    }
    s_hi = b;
    s_krem = TOPK - cum;
  }
  __syncthreads();
  const int hi = s_hi, krem = s_krem;
  hist[tid] = 0;
  __syncthreads();
  for (int i = tid; i < HIDDEN; i += 256) {
    unsigned int u = row[i];
    unsigned int key = u ^ ((u & 0x8000u) ? 0xFFFFu : 0x8000u);
    if ((int)(key >> 8) == hi) atomicAdd(&hist[key & 255u], 1u);
  }
  __syncthreads();
  if (tid == 0) {
    int cum = 0, b = 255;
    for (; b > 0; --b) {
      int c = (int)hist[b];
      if (cum + c >= krem) break;
      cum += c;
    }
    unsigned int key = ((unsigned int)hi << 8) | (unsigned int)b;
    unsigned int u = (key & 0x8000u) ? (key ^ 0x8000u) : (key ^ 0xFFFFu);
    thr[n] = bf2f((unsigned short)u);
  }
}

// ---------------------------- apply top-k mask -----------------------------
__global__ __launch_bounds__(256) void mask_kernel(
    const unsigned short* __restrict__ gb, const float* __restrict__ thr,
    unsigned short* __restrict__ tb) {
  const long long total = (long long)NTOK * HIDDEN;
  long long i = (long long)blockIdx.x * blockDim.x + threadIdx.x;
  long long stride = (long long)gridDim.x * blockDim.x;
  for (; i < total; i += stride) {
    const int n = (int)(i >> 13);  // / HIDDEN
    if (bf2f(gb[i]) < thr[n]) tb[i] = 0;
  }
}

// --------------------------- down GEMM (bf16 WMMA) -------------------------
// Grid: (D/128, N/64), block 256 (8 waves). Wave tile: 64 tokens x 16 d,
// K = HIDDEN (256 steps). Masked-t tile staged in LDS via async loads.
__global__ __launch_bounds__(256) void down_kernel(
    const unsigned short* __restrict__ tb,
    const unsigned short* __restrict__ wdb,
    float* __restrict__ out) {
  __shared__ __align__(16) unsigned short sA[2][64 * 32];

  const int lane   = threadIdx.x & 31;
  const int wave   = threadIdx.x >> 5;
  const int m0     = blockIdx.y << 6;
  const int d0     = (blockIdx.x << 7) + (wave << 4);
  const int r      = lane & 15;
  const int hiHalf = lane >> 4;

  const unsigned short* bBase = wdb + (size_t)(d0 + r) * HIDDEN + (hiHalf ? 16 : 0);
  const int aoff = hiHalf ? 16 : 0;

  v8f c[4] = {};

  stage_a_async(tb, HIDDEN, m0, 0, sA[0]);

  const int KT = HIDDEN / 32;
  for (int kt = 0; kt < KT; ++kt) {
    wait_async0();
    __syncthreads();
    if (kt + 1 < KT) stage_a_async(tb, HIDDEN, m0, (kt + 1) * 32, sA[(kt + 1) & 1]);

    const int k0 = kt * 32;
    v16bf b = load_frag(bBase + k0, 0, 8);
    const unsigned short* sa = sA[kt & 1];
#pragma unroll
    for (int tt = 0; tt < 4; ++tt) {
      v16bf a = load_frag(sa + (tt * 16 + r) * 32 + aoff, 0, 8);
      c[tt] = __builtin_amdgcn_wmma_f32_16x16x32_bf16(false, a, false, b, (short)0, c[tt], false, false);
    }
  }

  const int col = d0 + r;
#pragma unroll
  for (int tt = 0; tt < 4; ++tt) {
#pragma unroll
    for (int i = 0; i < 8; ++i) {
      const int row = m0 + tt * 16 + i + (hiHalf ? 8 : 0);
      out[(size_t)row * D_MODEL + col] = c[tt][i];
    }
  }
}

// ---------------------------------------------------------------------------
extern "C" void kernel_launch(void* const* d_in, const int* in_sizes, int n_in,
                              void* d_out, int out_size, void* d_ws, size_t ws_size,
                              hipStream_t stream) {
  const float* x      = (const float*)d_in[0];
  const float* w_up   = (const float*)d_in[1];
  const float* w_gate = (const float*)d_in[2];
  const float* w_down = (const float*)d_in[3];
  float* out = (float*)d_out;

  unsigned short* ws16 = (unsigned short*)d_ws;
  size_t o = 0;
  unsigned short* xb  = ws16 + o; o += (size_t)NTOK * D_MODEL;
  unsigned short* wub = ws16 + o; o += (size_t)HIDDEN * D_MODEL;
  unsigned short* wgb = ws16 + o; o += (size_t)HIDDEN * D_MODEL;
  unsigned short* wdb = ws16 + o; o += (size_t)D_MODEL * HIDDEN;
  unsigned short* gb  = ws16 + o; o += (size_t)NTOK * HIDDEN;
  unsigned short* tb  = ws16 + o; o += (size_t)NTOK * HIDDEN;
  float* thr = (float*)(ws16 + o);

  // 1) Convert operands to bf16 (weights then fit in 192 MB L2).
  cvt_bf16_kernel<<<4096, 256, 0, stream>>>(x,      xb,  (long long)NTOK * D_MODEL);
  cvt_bf16_kernel<<<4096, 256, 0, stream>>>(w_up,   wub, (long long)HIDDEN * D_MODEL);
  cvt_bf16_kernel<<<4096, 256, 0, stream>>>(w_gate, wgb, (long long)HIDDEN * D_MODEL);
  cvt_bf16_kernel<<<4096, 256, 0, stream>>>(w_down, wdb, (long long)D_MODEL * HIDDEN);

  // 2) Fused up+gate GEMM + silu (bf16 WMMA, fp32 accumulate).
  up_gate_kernel<<<dim3(HIDDEN / 128, NTOK / 64), 256, 0, stream>>>(xb, wub, wgb, gb, tb);

  // 3) Per-token top-k threshold (radix select on bf16 keys).
  topk_thresh_kernel<<<NTOK, 256, 0, stream>>>(gb, thr);

  // 4) Zero t below threshold.
  mask_kernel<<<4096, 256, 0, stream>>>(gb, thr, tb);

  // 5) Down GEMM -> fp32 output.
  down_kernel<<<dim3(D_MODEL / 128, NTOK / 64), 256, 0, stream>>>(tb, wdb, out);
}
